// LightweightGravitationalTransformer_32993938768210
// MI455X (gfx1250) — matrix operand order repaired and checked by reading, not compiled, not run on hardware
//
#include <hip/hip_runtime.h>
#include <cstdint>
#include <cstddef>

// ---------------- model constants ----------------
#define BB   4
#define SS   1024
#define DD   512
#define HH   8
#define DHD  64
#define LL   6
#define FFD  1024
#define VV   32000
#define BS   (BB*SS)          // 4096 token rows

typedef __attribute__((ext_vector_type(16))) __bf16 v16bf;
typedef __attribute__((ext_vector_type(8)))  float  v8f;
typedef __attribute__((ext_vector_type(4)))  unsigned int u32x4;
typedef __attribute__((ext_vector_type(8)))  int i32x8;
typedef __attribute__((ext_vector_type(4)))  int i32x4;

union FragAB { v16bf v; u32x4 q[2]; unsigned short s[16]; };

__device__ __forceinline__ unsigned short f2bf(float f) {
    unsigned int u = __float_as_uint(f);
    unsigned int r = (u + 0x7FFFu + ((u >> 16) & 1u)) >> 16;   // RNE
    return (unsigned short)r;
}
__device__ __forceinline__ float bf2f(unsigned short b) {
    return __uint_as_float(((unsigned int)b) << 16);
}
__device__ __forceinline__ float gelu_exact(float x) {
    return 0.5f * x * (1.0f + erff(x * 0.70710678118654752440f));
}
__device__ __forceinline__ unsigned lds_off(const void* p) {
    // generic pointers to LDS carry the LDS byte offset in addr[31:0] (ISA 10.2)
    return (unsigned)(uintptr_t)p;
}

// ---- Tensor Data Mover: 2D tile (tile_w elems x tile_h rows) global -> LDS ----
// D# built per CDNA5 ISA.
// dsz_code: 0=1B, 1=2B (bf16), 2=4B (f32), 3=8B
__device__ __forceinline__ void tdm_load_2d(void* ldsDst, const void* gsrc,
                                            unsigned dsz_code,
                                            unsigned tile_w, unsigned tile_h,
                                            unsigned tensor_w, unsigned tensor_h,
                                            unsigned long long stride0_elems) {
    unsigned long long ga = (unsigned long long)(uintptr_t)gsrc;
    u32x4 g0;
    g0[0] = 1u;                                              // count=1, user mode
    g0[1] = lds_off(ldsDst);                                 // lds_addr
    g0[2] = (unsigned)(ga & 0xFFFFFFFFu);                    // global_addr[31:0]
    g0[3] = (unsigned)((ga >> 32) & 0x01FFFFFFu) | (2u << 30); // addr[56:32] | type=2
    i32x8 g1;
    g1[0] = (int)(dsz_code << 16);                           // data_size
    g1[1] = (int)((tensor_w & 0xFFFFu) << 16);               // tensor_dim0[15:0]
    g1[2] = (int)(((tensor_w >> 16) & 0xFFFFu) |
                  ((tensor_h & 0xFFFFu) << 16));             // dim0[31:16] | dim1[15:0]
    g1[3] = (int)(((tensor_h >> 16) & 0xFFFFu) |
                  ((tile_w & 0xFFFFu) << 16));               // dim1[31:16] | tile_dim0
    g1[4] = (int)(tile_h & 0xFFFFu);                         // tile_dim1 (tile_dim2=0)
    g1[5] = (int)(stride0_elems & 0xFFFFFFFFu);              // dim0_stride[31:0]
    g1[6] = (int)((stride0_elems >> 32) & 0xFFFFu);          // dim0_stride[47:32]
    g1[7] = 0;
    i32x4 z4 = {0, 0, 0, 0};
#if defined(__clang_major__) && (__clang_major__ >= 23)
    i32x8 z8 = {0, 0, 0, 0, 0, 0, 0, 0};
    __builtin_amdgcn_tensor_load_to_lds(g0, g1, z4, z4, z8, 0);
#else
    __builtin_amdgcn_tensor_load_to_lds(g0, g1, z4, z4, 0);
#endif
}

// ---------------- embedding gather ----------------
__global__ void k_embed(const int* __restrict__ x, const float* __restrict__ emb,
                        float* __restrict__ h, unsigned short* __restrict__ hb) {
    int t = blockIdx.x;
    int tok = x[t];
    const float* src = emb + (size_t)tok * DD;
    for (int d = threadIdx.x; d < DD; d += blockDim.x) {
        float v = src[d];
        h [(size_t)t * DD + d] = v;
        hb[(size_t)t * DD + d] = f2bf(v);
    }
}

// ---------------- warped positions + inv distance ----------------
__global__ void k_pos(const float* __restrict__ pos, const float* __restrict__ curv,
                      float* __restrict__ pp, float* __restrict__ sq) {
    int s = blockIdx.x * blockDim.x + threadIdx.x;
    if (s >= SS) return;
    float c = curv[0];
    float acc = 0.f;
    for (int k = 0; k < 64; ++k) {
        float v0 = pos[(size_t)s * 64 + k];
        float v  = v0 * (1.0f + c * sinf(v0 * 0.1f));
        pp[(size_t)s * 64 + k] = v;
        acc += v * v;
    }
    sq[s] = acc;
}

__global__ void k_invd2(const float* __restrict__ pp, const float* __restrict__ sq,
                        float* __restrict__ inv) {
    __shared__ float pi[64];
    int i = blockIdx.y;
    int j = blockIdx.x * blockDim.x + threadIdx.x;
    if (threadIdx.x < 64) pi[threadIdx.x] = pp[(size_t)i * 64 + threadIdx.x];
    __syncthreads();
    float dot = 0.f;
    #pragma unroll 8
    for (int k = 0; k < 64; ++k) dot += pi[k] * pp[(size_t)j * 64 + k];
    float d2 = sq[i] + sq[j] - 2.0f * dot;
    d2 = fmaxf(d2, 1e-6f);
    inv[(size_t)i * SS + j] = 1.0f / d2;
}

// ---------------- fp32 -> bf16 transposed weight convert (K x N -> N x K) ----------------
__global__ void k_convt(const float* __restrict__ w, unsigned short* __restrict__ wt,
                        int K, int N) {
    long long t = (long long)blockIdx.x * blockDim.x + threadIdx.x;
    if (t >= (long long)K * N) return;
    int k = (int)(t / N), n = (int)(t % N);
    wt[(size_t)n * K + k] = f2bf(w[t]);
}

// ---------------- mass head: m = softplus(h @ W + b), N = 8 (VALU) ----------------
__global__ void k_mass(const float* __restrict__ h, const float* __restrict__ w,
                       const float* __restrict__ bias, float* __restrict__ m) {
    int t = blockIdx.x * blockDim.x + threadIdx.x;
    if (t >= BS * HH) return;
    int row = t >> 3, hh = t & 7;
    const float* hr = h + (size_t)row * DD;
    float s = bias[hh];
    #pragma unroll 8
    for (int d = 0; d < DD; ++d) s += hr[d] * w[(size_t)d * HH + hh];
    m[t] = (s > 20.0f) ? s : log1pf(expf(s));
}

// ---------------- bf16 WMMA GEMM, TDM double-buffered through LDS ----------------
// C = A(MxK) * Wt(NxK)^T + bias.  Block tile 64x256, 8 waves (2x4), wave tile 32x64
// (2 M-frags x 4 N-frags = 8 WMMA per K-step per wave).
// OUTMODE: 0 = fp32 row-major, 1 = bf16 row-major, 2 = bf16 transposed (ld = M)
template<int ACT, int OUTMODE>
__global__ void k_gemm(const unsigned short* __restrict__ A,
                       const unsigned short* __restrict__ Wt,
                       const float* __restrict__ bias,
                       float* __restrict__ Cf, unsigned short* __restrict__ Cb,
                       int M, int N, int K) {
    __shared__ unsigned short sA[2][64 * 32];      //  4 KB per buffer
    __shared__ unsigned short sB[2][256 * 32];     // 16 KB per buffer

    const int lane  = threadIdx.x & 31;
    const int wid   = threadIdx.x >> 5;
    const int wm    = wid >> 2;        // 2 waves along M (32 rows each)
    const int wn    = wid & 3;         // 4 waves along N (64 cols each)
    const int row0  = blockIdx.y * 64;
    const int col0  = blockIdx.x * 256;
    const int r     = lane & 15;
    const int khalf = lane >> 4;

    v8f acc[2][4];
    #pragma unroll
    for (int mm = 0; mm < 2; ++mm)
        #pragma unroll
        for (int f = 0; f < 4; ++f)
            #pragma unroll
            for (int e = 0; e < 8; ++e) acc[mm][f][e] = 0.0f;

    const int nIter = K >> 5;

    if (wid == 0) {       // wave 0 drives the Tensor Data Mover
        tdm_load_2d(&sA[0][0], A  + (size_t)row0 * K, 1, 32, 64,  K, M - row0, K);
        tdm_load_2d(&sB[0][0], Wt + (size_t)col0 * K, 1, 32, 256, K, N - col0, K);
    }

    for (int it = 0; it < nIter; ++it) {
        const int p = it & 1;
        if (wid == 0) {
            if (it + 1 < nIter) {
                const int k0n = (it + 1) << 5;
                tdm_load_2d(&sA[p ^ 1][0], A  + (size_t)row0 * K + k0n, 1, 32, 64,  K, M - row0, K);
                tdm_load_2d(&sB[p ^ 1][0], Wt + (size_t)col0 * K + k0n, 1, 32, 256, K, N - col0, K);
                __builtin_amdgcn_s_wait_tensorcnt(2);   // current tile done, next in flight
            } else {
                __builtin_amdgcn_s_wait_tensorcnt(0);
            }
        }
        __syncthreads();

        FragAB a[2], b;
        #pragma unroll
        for (int mm = 0; mm < 2; ++mm) {
            const unsigned short* ar = &sA[p][(wm * 32 + mm * 16 + r) * 32 + khalf * 8];
            a[mm].q[0] = *(const u32x4*)(ar);
            a[mm].q[1] = *(const u32x4*)(ar + 16);
        }
        #pragma unroll
        for (int f = 0; f < 4; ++f) {
            const unsigned short* br = &sB[p][(wn * 64 + f * 16 + r) * 32 + khalf * 8];
            b.q[0] = *(const u32x4*)(br);
            b.q[1] = *(const u32x4*)(br + 16);
            #pragma unroll
            for (int mm = 0; mm < 2; ++mm)
                acc[mm][f] = __builtin_amdgcn_wmma_f32_16x16x32_bf16(
                                 false, a[mm].v, false, b.v, (short)0, acc[mm][f], false, false);
        }
        __syncthreads();   // everyone done reading p before it gets overwritten
    }

    #pragma unroll
    for (int mm = 0; mm < 2; ++mm) {
        const int wrow0 = row0 + wm * 32 + mm * 16;
        #pragma unroll
        for (int f = 0; f < 4; ++f) {
            int col = col0 + wn * 64 + f * 16 + r;
            float bv = bias ? bias[col] : 0.0f;
            if (OUTMODE == 2) {
                unsigned short tv[8];
                #pragma unroll
                for (int v = 0; v < 8; ++v) {
                    float val = acc[mm][f][v] + bv;
                    if (ACT == 1) val = gelu_exact(val);
                    tv[v] = f2bf(val);
                }
                u32x4 pk;
                pk[0] = (unsigned)tv[0] | ((unsigned)tv[1] << 16);
                pk[1] = (unsigned)tv[2] | ((unsigned)tv[3] << 16);
                pk[2] = (unsigned)tv[4] | ((unsigned)tv[5] << 16);
                pk[3] = (unsigned)tv[6] | ((unsigned)tv[7] << 16);
                *(u32x4*)(Cb + (size_t)col * M + wrow0 + khalf * 8) = pk;
            } else {
                #pragma unroll
                for (int v = 0; v < 8; ++v) {
                    int row = wrow0 + khalf * 8 + v;
                    float val = acc[mm][f][v] + bv;
                    if (ACT == 1) val = gelu_exact(val);
                    if (OUTMODE == 0) Cf[(size_t)row * N + col] = val;
                    else              Cb[(size_t)row * N + col] = f2bf(val);
                }
            }
        }
    }
}

// ---------------- gravitational attention: o = (sum_j F_ij v_j) / (sum_j F_ij) ----------------
// One wave per 16-row i-tile per (b, head). Per j-tile, the TDM stages three tiles into
// LDS double buffers: v (64x32 bf16), the inv_d2 slab (16x32 f32, stride S), and the
// m_j gather (32 rows x 1 elem, stride H). F is built from LDS, packed to bf16 A-frags,
// and WMMA-accumulated against the v-tile; one linear normalization at the end.
__global__ void k_attn(const float* __restrict__ m,            // (B*S, H)
                       const unsigned short* __restrict__ vt,  // (H*64, B*S) bf16
                       const float* __restrict__ inv_d2,       // (S, S)
                       const float* __restrict__ g,            // (H) for this layer
                       unsigned short* __restrict__ ob) {      // (B*S, D) bf16
    __shared__ unsigned short sV[2][64 * 32];    // 4 KB per buffer
    __shared__ float sD2[2][16 * 32];            // 2 KB per buffer
    __shared__ float sMj[2][32];
    __shared__ float s_rs[16];

    const int lane  = threadIdx.x & 31;
    const int i0    = blockIdx.x * 16;
    const int hh    = blockIdx.y;
    const int b     = blockIdx.z;
    const int r     = lane & 15;
    const int khalf = lane >> 4;

    const float gh  = g[hh];
    const float mi  = m[((size_t)(b * SS + i0 + r)) * HH + hh];
    const float* dbase = inv_d2 + (size_t)i0 * SS;
    const float* mbase = m + (size_t)b * SS * HH + hh;
    const unsigned short* vbase = vt + ((size_t)hh * DHD) * BS + (size_t)b * SS;

    v8f acc[4];
    #pragma unroll
    for (int f = 0; f < 4; ++f)
        #pragma unroll
        for (int e = 0; e < 8; ++e) acc[f][e] = 0.0f;

    float rowsum = 0.0f;
    const float gm = gh * mi;
    const int nIter = SS >> 5;      // 32 j-tiles

    // prologue: j-tile 0 (v, d2 slab, m_j gather)
    tdm_load_2d(&sV[0][0],  vbase, 1, 32, 64, 32, 64, BS);
    tdm_load_2d(&sD2[0][0], dbase, 2, 32, 16, SS, 16, SS);
    tdm_load_2d(&sMj[0][0], mbase, 2, 1, 32, 1, 32, HH);

    for (int it = 0; it < nIter; ++it) {
        const int p  = it & 1;
        const int j0 = it << 5;
        if (it + 1 < nIter) {
            const int jn = j0 + 32;
            tdm_load_2d(&sV[p ^ 1][0],  vbase + jn, 1, 32, 64, 32, 64, BS);
            tdm_load_2d(&sD2[p ^ 1][0], dbase + jn, 2, 32, 16, SS, 16, SS);
            tdm_load_2d(&sMj[p ^ 1][0], mbase + (size_t)jn * HH, 2, 1, 32, 1, 32, HH);
            __builtin_amdgcn_s_wait_tensorcnt(3);   // current 3 tiles resident
        } else {
            __builtin_amdgcn_s_wait_tensorcnt(0);
        }

        FragAB a;
        #pragma unroll
        for (int t = 0; t < 16; ++t) {
            int jrel = (t < 8) ? (khalf * 8 + t) : (16 + khalf * 8 + (t - 8));
            float F  = fminf(gm * sMj[p][jrel] * sD2[p][r * 32 + jrel], 50.0f);
            unsigned short us = f2bf(F);
            a.s[t] = us;
            rowsum += bf2f(us);   // sum the rounded value: numerator/denominator match
        }
        #pragma unroll
        for (int f = 0; f < 4; ++f) {
            const unsigned short* br = &sV[p][(f * 16 + r) * 32 + khalf * 8];
            FragAB bf;
            bf.q[0] = *(const u32x4*)(br);
            bf.q[1] = *(const u32x4*)(br + 16);
            acc[f] = __builtin_amdgcn_wmma_f32_16x16x32_bf16(
                         false, a.v, false, bf.v, (short)0, acc[f], false, false);
        }
    }

    rowsum += __shfl_xor(rowsum, 16, 32);
    if (lane < 16) s_rs[r] = rowsum;
    __syncthreads();

    #pragma unroll
    for (int f = 0; f < 4; ++f) {
        #pragma unroll
        for (int v = 0; v < 8; ++v) {
            int rl  = khalf * 8 + v;
            float o = acc[f][v] / s_rs[rl];
            ob[((size_t)(b * SS + i0 + rl)) * DD + hh * DHD + f * 16 + r] = f2bf(o);
        }
    }
}

// ---------------- residual add + LayerNorm (writes fp32 master + bf16 copy) ----------------
__global__ void k_add_ln(float* __restrict__ h, const float* __restrict__ o,
                         const float* __restrict__ sc, const float* __restrict__ bt,
                         unsigned short* __restrict__ hb, int addFlag) {
    __shared__ float red[8];
    int row = blockIdx.x;
    float xv[2];
    #pragma unroll
    for (int e = 0; e < 2; ++e) {
        int d = threadIdx.x + e * 256;
        float v = h[(size_t)row * DD + d];
        if (addFlag) v += o[(size_t)row * DD + d];
        xv[e] = v;
    }
    float s = xv[0] + xv[1];
    for (int off = 16; off > 0; off >>= 1) s += __shfl_xor(s, off, 32);
    if ((threadIdx.x & 31) == 0) red[threadIdx.x >> 5] = s;
    __syncthreads();
    float tot = 0.f;
    #pragma unroll
    for (int i = 0; i < 8; ++i) tot += red[i];
    float mu = tot * (1.0f / DD);
    __syncthreads();
    float v2 = (xv[0] - mu) * (xv[0] - mu) + (xv[1] - mu) * (xv[1] - mu);
    for (int off = 16; off > 0; off >>= 1) v2 += __shfl_xor(v2, off, 32);
    if ((threadIdx.x & 31) == 0) red[threadIdx.x >> 5] = v2;
    __syncthreads();
    float vt = 0.f;
    #pragma unroll
    for (int i = 0; i < 8; ++i) vt += red[i];
    float rstd = rsqrtf(vt * (1.0f / DD) + 1e-5f);
    #pragma unroll
    for (int e = 0; e < 2; ++e) {
        int d = threadIdx.x + e * 256;
        float y = (xv[e] - mu) * rstd * sc[d] + bt[d];
        h [(size_t)row * DD + d] = y;
        hb[(size_t)row * DD + d] = f2bf(y);
    }
}

// ---------------- host orchestration ----------------
extern "C" void kernel_launch(void* const* d_in, const int* in_sizes, int n_in,
                              void* d_out, int out_size, void* d_ws, size_t ws_size,
                              hipStream_t stream) {
    const int*   x       = (const int*)  d_in[0];
    const float* emb     = (const float*)d_in[1];
    const float* pos     = (const float*)d_in[2];
    const float* curv    = (const float*)d_in[3];
    const float* g_heads = (const float*)d_in[4];
    const float* v_w     = (const float*)d_in[5];
    const float* v_b     = (const float*)d_in[6];
    const float* mass_w  = (const float*)d_in[7];
    const float* mass_b  = (const float*)d_in[8];
    const float* out_w   = (const float*)d_in[9];
    const float* out_b   = (const float*)d_in[10];
    const float* ffn_w1  = (const float*)d_in[11];
    const float* ffn_b1  = (const float*)d_in[12];
    const float* ffn_w2  = (const float*)d_in[13];
    const float* ffn_b2  = (const float*)d_in[14];
    const float* ln1_s   = (const float*)d_in[15];
    const float* ln1_b   = (const float*)d_in[16];
    const float* ln2_s   = (const float*)d_in[17];
    const float* ln2_b   = (const float*)d_in[18];
    const float* fin_s   = (const float*)d_in[19];
    const float* fin_b   = (const float*)d_in[20];
    const float* head_w  = (const float*)d_in[21];
    const float* head_b  = (const float*)d_in[22];

    char* ws = (char*)d_ws;
    size_t off = 0;
    auto take = [&](size_t bytes) -> void* {
        void* p = ws + off;
        off = (off + bytes + 255) & ~(size_t)255;
        return p;
    };
    float*          Hbuf = (float*)         take((size_t)BS * DD * 4);
    unsigned short* HB   = (unsigned short*)take((size_t)BS * DD * 2);
    float*          TMP  = (float*)         take((size_t)BS * DD * 4);
    unsigned short* OB   = (unsigned short*)take((size_t)BS * DD * 2);
    unsigned short* FB   = (unsigned short*)take((size_t)BS * FFD * 2);
    unsigned short* VT   = (unsigned short*)take((size_t)BS * DD * 2);
    float*          Mm   = (float*)         take((size_t)BS * HH * 4);
    float*          PP   = (float*)         take((size_t)SS * 64 * 4);
    float*          SQ   = (float*)         take((size_t)SS * 4);
    float*          D2   = (float*)         take((size_t)SS * SS * 4);
    unsigned short* VWt  = (unsigned short*)take((size_t)LL * DD * DD * 2);
    unsigned short* OWt  = (unsigned short*)take((size_t)LL * DD * DD * 2);
    unsigned short* F1t  = (unsigned short*)take((size_t)LL * DD * FFD * 2);
    unsigned short* F2t  = (unsigned short*)take((size_t)LL * FFD * DD * 2);
    unsigned short* HWt  = (unsigned short*)take((size_t)DD * VV * 2);

    auto cvt = [&](const float* w, unsigned short* wt, int K, int N) {
        long long total = (long long)K * N;
        int blocks = (int)((total + 255) / 256);
        k_convt<<<blocks, 256, 0, stream>>>(w, wt, K, N);
    };
    for (int l = 0; l < LL; ++l) {
        cvt(v_w    + (size_t)l * DD * DD,  VWt + (size_t)l * DD * DD,  DD,  DD);
        cvt(out_w  + (size_t)l * DD * DD,  OWt + (size_t)l * DD * DD,  DD,  DD);
        cvt(ffn_w1 + (size_t)l * DD * FFD, F1t + (size_t)l * DD * FFD, DD,  FFD);
        cvt(ffn_w2 + (size_t)l * FFD * DD, F2t + (size_t)l * FFD * DD, FFD, DD);
    }
    cvt(head_w, HWt, DD, VV);

    k_embed<<<BS, 256, 0, stream>>>(x, emb, Hbuf, HB);
    k_pos  <<<(SS + 255) / 256, 256, 0, stream>>>(pos, curv, PP, SQ);
    k_invd2<<<dim3(SS / 256, SS), 256, 0, stream>>>(PP, SQ, D2);

    for (int l = 0; l < LL; ++l) {
        k_mass<<<(BS * HH + 255) / 256, 256, 0, stream>>>(
            Hbuf, mass_w + (size_t)l * DD * HH, mass_b + (size_t)l * HH, Mm);

        // v = h @ v_w + v_b  -> transposed bf16 (H*64, B*S)
        k_gemm<0, 2><<<dim3(DD / 256, BS / 64), 256, 0, stream>>>(
            HB, VWt + (size_t)l * DD * DD, v_b + (size_t)l * DD,
            nullptr, VT, BS, DD, DD);

        k_attn<<<dim3(SS / 16, HH, BB), 32, 0, stream>>>(
            Mm, VT, D2, g_heads + (size_t)l * HH, OB);

        // out projection -> fp32 TMP
        k_gemm<0, 0><<<dim3(DD / 256, BS / 64), 256, 0, stream>>>(
            OB, OWt + (size_t)l * DD * DD, out_b + (size_t)l * DD,
            TMP, nullptr, BS, DD, DD);

        k_add_ln<<<BS, 256, 0, stream>>>(Hbuf, TMP,
            ln1_s + (size_t)l * DD, ln1_b + (size_t)l * DD, HB, 1);

        // FFN1 (gelu) -> bf16 FB
        k_gemm<1, 1><<<dim3(FFD / 256, BS / 64), 256, 0, stream>>>(
            HB, F1t + (size_t)l * DD * FFD, ffn_b1 + (size_t)l * FFD,
            nullptr, FB, BS, FFD, DD);

        // FFN2 -> fp32 TMP
        k_gemm<0, 0><<<dim3(DD / 256, BS / 64), 256, 0, stream>>>(
            FB, F2t + (size_t)l * FFD * DD, ffn_b2 + (size_t)l * DD,
            TMP, nullptr, BS, DD, FFD);

        k_add_ln<<<BS, 256, 0, stream>>>(Hbuf, TMP,
            ln2_s + (size_t)l * DD, ln2_b + (size_t)l * DD, HB, 1);
    }

    k_add_ln<<<BS, 256, 0, stream>>>(Hbuf, nullptr, fin_s, fin_b, HB, 0);

    // head: logits fp32 straight into d_out (bound by the 524 MB write)
    k_gemm<0, 0><<<dim3(VV / 256, BS / 64), 256, 0, stream>>>(
        HB, HWt, head_b, (float*)d_out, nullptr, BS, VV, DD);
}
